// UnsupervisedModel_19911468384614
// MI455X (gfx1250) — matrix operand order repaired and checked by reading, not compiled, not run on hardware
//
#include <hip/hip_runtime.h>

// ---------------------------------------------------------------------------
// GCN stack for MI455X (gfx1250, wave32, WMMA + TDM).
// Per layer: H = X @ W via v_wmma_f32_16x16x32_f16 (f16 operands staged once,
// W^T pulled into LDS by the Tensor Data Mover), then edge scatter (f32
// atomics, b128 gathers, software prefetch), then fused self-loop+bias+ReLU.
// GEMM shapes (K, OUTD) are compile-time so the K loop fully unrolls and all
// addresses fold into immediate offsets.
// ---------------------------------------------------------------------------

typedef __attribute__((ext_vector_type(4)))  _Float16     v4h;
typedef __attribute__((ext_vector_type(8)))  _Float16     v8h;
typedef __attribute__((ext_vector_type(16))) _Float16     v16h;
typedef __attribute__((ext_vector_type(8)))  float        v8f;
typedef __attribute__((ext_vector_type(4)))  unsigned int u32x4;
typedef __attribute__((ext_vector_type(8)))  int          i32x8;
typedef __attribute__((ext_vector_type(4)))  int          i32x4;

#if defined(__gfx1250__) && __has_builtin(__builtin_amdgcn_tensor_load_to_lds) && \
    __has_builtin(__builtin_amdgcn_s_wait_tensorcnt)
#define USE_TDM 1
#else
#define USE_TDM 0
#endif

// ---- elementwise utility kernels ------------------------------------------

__global__ void k_fill4(float4* __restrict__ p, float v, int n4) {
  int i = blockIdx.x * blockDim.x + threadIdx.x;
  if (i < n4) p[i] = make_float4(v, v, v, v);
}

__global__ void k_deg(const int* __restrict__ dst, float* __restrict__ deg, int E) {
  int i = blockIdx.x * blockDim.x + threadIdx.x;
  if (i < E) atomicAdd(&deg[dst[i]], 1.0f);
}

__global__ void k_norm(const float* __restrict__ deg, float* __restrict__ isd,
                       float* __restrict__ invd, int n) {
  int i = blockIdx.x * blockDim.x + threadIdx.x;
  if (i < n) {
    float d = deg[i];
    isd[i]  = rsqrtf(d);
    invd[i] = 1.0f / d;
  }
}

// f32 -> f16, 4 elements per thread
__global__ void k_cvt(const float4* __restrict__ X, v4h* __restrict__ Xh, int total4) {
  int i = blockIdx.x * blockDim.x + threadIdx.x;
  if (i < total4) {
    const float4 v = X[i];
    v4h h;
    h.x = (_Float16)v.x; h.y = (_Float16)v.y;
    h.z = (_Float16)v.z; h.w = (_Float16)v.w;
    Xh[i] = h;
  }
}

// W[K][outd] (f32) -> Wt[outd][K] (f16)
__global__ void k_wt(const float* __restrict__ W, _Float16* __restrict__ Wt,
                     int K, int outd) {
  int idx = blockIdx.x * blockDim.x + threadIdx.x;
  if (idx < K * outd) {
    const int ncol = idx / K;
    const int k    = idx - ncol * K;
    Wt[idx] = (_Float16)W[(size_t)k * outd + ncol];
  }
}

// ---- dense GEMM: H[n x OUTD] = Xh[n x K] * W (Wt = W^T, f16) ---------------
// One wave computes a 16x64 output strip (4 WMMA tiles). Wt is staged into
// LDS by the Tensor Data Mover (one wave issues TENSOR_LOAD_TO_LDS, waits on
// TENSORcnt, block barriers). A: 2x global b128 per K-step, reused by all 4
// column tiles; B: 8x ds_load_b128 per K-step. K/OUTD are compile-time, so
// the K loop fully unrolls and loads/stores use immediate offsets.

template <int K, int OUTD>
__launch_bounds__(256)
__global__ void k_gemm_wmma(const _Float16* __restrict__ Xh,
                            const _Float16* __restrict__ Wt,   // [OUTD][K]
                            float* __restrict__ H, int n) {
  __align__(32) __shared__ _Float16 sW[OUTD * K];
  const int tid = threadIdx.x;

#if USE_TDM
  if (tid < 32) {
    // D# for a 1-D byte tile: Wt[0 .. OUTD*K*2) -> LDS sW.
    const unsigned long long ga = (unsigned long long)(const void*)Wt;
    const unsigned lds = (unsigned)(unsigned long long)(void*)sW;
    const unsigned B = (unsigned)(OUTD * K) * 2u;      // total bytes (<= 32768)

    u32x4 g0;
    g0.x = 1u;                                         // count=1, user mode
    g0.y = lds;                                        // lds_addr
    g0.z = (unsigned)(ga & 0xFFFFFFFFu);               // global_addr[31:0]
    g0.w = (unsigned)((ga >> 32) & 0x01FFFFFFu) | (2u << 30);  // [56:32]|type=2

    i32x8 g1;
    g1[0] = 0;                                         // wg_mask=0, data_size=1B
    g1[1] = (int)((B & 0xFFFFu) << 16);                // tensor_dim0 lo16
    g1[2] = (int)((B >> 16) & 0xFFFFu) | (1 << 16);    // dim0 hi16 | dim1=1 lo
    g1[3] = (int)((B & 0xFFFFu) << 16);                // dim1 hi=0 | tile_dim0=B
    g1[4] = 1;                                         // tile_dim1=1, tile_dim2=0
    g1[5] = (int)B;                                    // dim0_stride lo32
    g1[6] = 0;
    g1[7] = 0;

    const i32x4 z4 = {0, 0, 0, 0};
#if __clang_major__ >= 23
    const i32x8 z8 = {0, 0, 0, 0, 0, 0, 0, 0};
    __builtin_amdgcn_tensor_load_to_lds(g0, g1, z4, z4, z8, 0);
#else
    __builtin_amdgcn_tensor_load_to_lds(g0, g1, z4, z4, 0);
#endif
    __builtin_amdgcn_s_wait_tensorcnt(0);
  }
#else
  {
    const int total8 = (OUTD * K) >> 3;
    const v8h* __restrict__ g8 = (const v8h*)Wt;
    v8h* s8 = (v8h*)sW;
    for (int i = tid; i < total8; i += 256) s8[i] = g8[i];
  }
#endif
  __syncthreads();

  constexpr int colBlocks = OUTD >> 6;           // 64 columns per wave
  const int wave = blockIdx.x * 8 + (tid >> 5);
  const int lane = tid & 31;
  const int totalW = (n >> 4) * colBlocks;
  if (wave >= totalW) return;

  const int tm = wave / colBlocks;
  const int cb = wave - tm * colBlocks;
  const int r0 = tm << 4;
  const int c0 = cb << 6;
  const int g  = lane >> 4;                      // half-wave group
  const int m  = lane & 15;                      // A row / C column in tile

  const _Float16* __restrict__ Arow = Xh + (size_t)(r0 + m) * K + (g << 4);
  const _Float16* __restrict__ Brow = sW + (c0 + m) * K + (g << 4);

  v8f acc[4] = {};
#pragma unroll
  for (int kk = 0; kk < K; kk += 32) {
    // A 16x32 f16 layout (contiguous, matches B): elem e -> K = g*16+e;
    // two contiguous 16B loads.
    const v8h a0 = *(const v8h*)(Arow + kk);
    const v8h a1 = *(const v8h*)(Arow + kk + 8);
    const v16h a = __builtin_shufflevector(a0, a1,
        0, 1, 2, 3, 4, 5, 6, 7, 8, 9, 10, 11, 12, 13, 14, 15);

    // Preload all 4 B tiles, then 4 WMMAs back-to-back.
    v16h b[4];
#pragma unroll
    for (int t = 0; t < 4; ++t)
      b[t] = *(const v16h*)(Brow + t * (K << 4) + kk);
#pragma unroll
    for (int t = 0; t < 4; ++t)
      acc[t] = __builtin_amdgcn_wmma_f32_16x16x32_f16(
          false, a, false, b[t], (short)0, acc[t], false, false);
  }

  // C/D layout: VGPR r -> M = r + 8*g, N = lane&15
#pragma unroll
  for (int t = 0; t < 4; ++t) {
    float* __restrict__ base = H + (size_t)(r0 + (g << 3)) * OUTD + c0 + (t << 4) + m;
#pragma unroll
    for (int r = 0; r < 8; ++r)
      base[r * OUTD] = acc[t][r];
  }
}

// ---- edge aggregation: agg[dst] += H[src] * isd[src]*isd[dst] --------------
// One wave per edge; each lane owns VEC consecutive features (b128/b64 gather
// + VEC f32 atomics). Prefetch a future edge's source row to hide latency.

template <int VEC>
__launch_bounds__(256)
__global__ void k_scatter(const int* __restrict__ src, const int* __restrict__ dst,
                          const float* __restrict__ isd, const float* __restrict__ H,
                          float* __restrict__ agg, int E) {
  typedef __attribute__((ext_vector_type(VEC))) float vf;
  constexpr int OUTD = VEC * 32;
  const int e    = blockIdx.x * 8 + (threadIdx.x >> 5);
  const int lane = threadIdx.x & 31;
  if (e >= E) return;

  const int ep = e + 4096;                       // software prefetch distance
  if (ep < E)
    __builtin_prefetch(H + (size_t)src[ep] * OUTD + lane * VEC, 0, 0);

  const int s = src[e], d = dst[e];
  const float nrm = isd[s] * isd[d];
  const vf h = *(const vf*)(H + (size_t)s * OUTD + lane * VEC);
  float* __restrict__ ad = agg + (size_t)d * OUTD + lane * VEC;
#pragma unroll
  for (int i = 0; i < VEC; ++i) atomicAdd(&ad[i], h[i] * nrm);
}

// ---- finalize: out += H * invdeg (self loop) + bias; optional ReLU ---------

__global__ void k_finalize4(float4* __restrict__ out, const float4* __restrict__ H,
                            const float* __restrict__ invd, const float* __restrict__ bias,
                            int total4, int outd_mask, int outd_shift, int relu) {
  int i4 = blockIdx.x * blockDim.x + threadIdx.x;
  if (i4 >= total4) return;
  const int base = i4 << 2;
  const int node = base >> outd_shift;
  const int f    = base & outd_mask;
  const float id = invd[node];
  float4 o = out[i4];
  const float4 h = H[i4];
  const float4 b = *(const float4*)(bias + f);
  o.x += h.x * id + b.x;
  o.y += h.y * id + b.y;
  o.z += h.z * id + b.z;
  o.w += h.w * id + b.w;
  if (relu) {
    o.x = fmaxf(o.x, 0.0f); o.y = fmaxf(o.y, 0.0f);
    o.z = fmaxf(o.z, 0.0f); o.w = fmaxf(o.w, 0.0f);
  }
  out[i4] = o;
}

// ---------------------------------------------------------------------------

extern "C" void kernel_launch(void* const* d_in, const int* in_sizes, int n_in,
                              void* d_out, int out_size, void* d_ws, size_t ws_size,
                              hipStream_t stream) {
  const float* x  = (const float*)d_in[0];
  const float* W1 = (const float*)d_in[1];
  const float* b1 = (const float*)d_in[2];
  const float* W2 = (const float*)d_in[3];
  const float* b2 = (const float*)d_in[4];
  const float* W3 = (const float*)d_in[5];
  const float* b3 = (const float*)d_in[6];
  const float* W4 = (const float*)d_in[7];
  const float* b4 = (const float*)d_in[8];
  const int* edge_index = (const int*)d_in[9];

  const int DIN = 128, DPROJ = 128, DHID = 64;
  const int N = in_sizes[0] / DIN;
  const int E = in_sizes[9] / 2;
  const int* srcp = edge_index;
  const int* dstp = edge_index + E;

  // workspace carve-out
  float* ws   = (float*)d_ws;
  float* deg  = ws;                              // N
  float* isd  = deg  + N;                        // N
  float* invd = isd  + N;                        // N
  float* H    = invd + N;                        // N*128 f32 (pre-agg x@W)
  float* buf0 = H    + (size_t)N * 128;          // N*128 f32
  float* buf1 = buf0 + (size_t)N * 128;          // N*64  f32
  _Float16* Xh = (_Float16*)(buf1 + (size_t)N * 64);  // N*128 f16
  _Float16* Wt = Xh + (size_t)N * 128;           // 128*128 f16 (W^T staging)

  const int T = 256;

  // degrees (with self loop), normalization factors
  k_fill4<<<((N / 4) + T - 1) / T, T, 0, stream>>>((float4*)deg, 1.0f, N / 4);
  k_deg<<<(E + T - 1) / T, T, 0, stream>>>(dstp, deg, E);
  k_norm<<<(N + T - 1) / T, T, 0, stream>>>(deg, isd, invd, N);

  auto layer = [&](const float* in, int K, const float* Wl, const float* bl,
                   int outd, float* out, int relu, int outd_shift) {
    const int total  = N * outd;
    const int total4 = total >> 2;
    const int inTot4 = (N * K) >> 2;
    // stage f16 operands
    k_cvt<<<(inTot4 + T - 1) / T, T, 0, stream>>>((const float4*)in, (v4h*)Xh, inTot4);
    k_wt<<<(K * outd + T - 1) / T, T, 0, stream>>>(Wl, Wt, K, outd);
    // dense GEMM (WMMA), shape-specialized
    const int waves  = (N >> 4) * (outd >> 6);
    const int blocks = (waves + 7) / 8;
    if (K == 128 && outd == 128)
      k_gemm_wmma<128, 128><<<blocks, 256, 0, stream>>>(Xh, Wt, H, N);
    else if (K == 128 && outd == 64)
      k_gemm_wmma<128, 64><<<blocks, 256, 0, stream>>>(Xh, Wt, H, N);
    else
      k_gemm_wmma<64, 128><<<blocks, 256, 0, stream>>>(Xh, Wt, H, N);
    // neighbor aggregation
    k_fill4<<<(total4 + T - 1) / T, T, 0, stream>>>((float4*)out, 0.0f, total4);
    if (outd == 128)
      k_scatter<4><<<(E + 7) / 8, 256, 0, stream>>>(srcp, dstp, isd, H, out, E);
    else
      k_scatter<2><<<(E + 7) / 8, 256, 0, stream>>>(srcp, dstp, isd, H, out, E);
    // self loop + bias (+ ReLU)
    k_finalize4<<<(total4 + T - 1) / T, T, 0, stream>>>(
        (float4*)out, (const float4*)H, invd, bl, total4, outd - 1, outd_shift, relu);
  };

  // encoder
  layer(x,    DIN,   W1, b1, DPROJ, buf0, /*relu=*/1, 7);
  layer(buf0, DPROJ, W2, b2, DHID,  buf1, /*relu=*/0, 6);
  // contra head
  layer(buf1, DHID,  W3, b3, DPROJ, buf0, /*relu=*/1, 7);
  layer(buf0, DPROJ, W4, b4, DIN, (float*)d_out, /*relu=*/0, 7);
}
